// SpatioTemporalSelfAttention_1992864825801
// MI455X (gfx1250) — matrix operand order repaired and checked by reading, not compiled, not run on hardware
//
#include <hip/hip_runtime.h>
#include <hip/hip_bf16.h>
#include <math.h>

typedef unsigned short ushort_t;
typedef unsigned int   uint32;
typedef __attribute__((ext_vector_type(16))) __bf16 v16bf;
typedef __attribute__((ext_vector_type(8)))  float  v8f;
typedef __attribute__((ext_vector_type(4)))  int    v4i;

#define LN_EPS 1e-5f
#define BM 128
#define BN 128
#define BK 32
#define LDP (BK + 8)   // LDS row pitch in ushorts: 40 -> 80 bytes, 16B multiple

// ---- CDNA5 async global->LDS path (ASYNCcnt), with synchronous fallback ----
#if __has_builtin(__builtin_amdgcn_global_load_async_to_lds_b128) && \
    __has_builtin(__builtin_amdgcn_s_wait_asynccnt)
#define USE_ASYNC 1
#else
#define USE_ASYNC 0
#endif

typedef __attribute__((address_space(1))) v4i gv4i;   // global
typedef __attribute__((address_space(3))) v4i lv4i;   // LDS

__device__ __forceinline__ void copy16_g2l(const ushort_t* g, ushort_t* l) {
#if USE_ASYNC
    __builtin_amdgcn_global_load_async_to_lds_b128((gv4i*)g, (lv4i*)l, 0, 0);
#else
    *(uint4*)l = *(const uint4*)g;
#endif
}
__device__ __forceinline__ void async_wait_all() {
#if USE_ASYNC
    __builtin_amdgcn_s_wait_asynccnt(0);
#endif
}

__device__ __forceinline__ ushort_t f2bf(float f) {
    uint32 u = __float_as_uint(f);
    u += 0x7FFFu + ((u >> 16) & 1u);          // round-to-nearest-even
    return (ushort_t)(u >> 16);
}

// ---------------- fp32 -> bf16 weight conversion ----------------
__global__ void cvt_f32_bf16_kernel(const float* __restrict__ in,
                                    ushort_t* __restrict__ out, int n) {
    int i = blockIdx.x * blockDim.x + threadIdx.x;
    if (i < n) out[i] = f2bf(in[i]);
}

// ---------------- LayerNorm statistics (per token) ----------------
__global__ void ln_stats_kernel(const float* __restrict__ x,
                                float* __restrict__ stats, int C, int Ntok) {
    int token = blockIdx.x * blockDim.x + threadIdx.x;   // 0 .. B*N-1
    int b = token / Ntok, n = token % Ntok;
    const float* p = x + (long)b * C * Ntok + n;
    float s = 0.f, s2 = 0.f;
    for (int c = 0; c < C; ++c) {
        float v = p[(long)c * Ntok];
        s += v; s2 += v * v;
    }
    float mu  = s / (float)C;
    float var = s2 / (float)C - mu * mu;
    stats[2 * token]     = mu;
    stats[2 * token + 1] = rsqrtf(var + LN_EPS);
}

// ---------------- LN apply + transpose BCN -> BNC (bf16) ----------------
__global__ void ln_apply_kernel(const float* __restrict__ x,
                                const float* __restrict__ stats,
                                const float* __restrict__ gamma,
                                const float* __restrict__ beta,
                                ushort_t* __restrict__ Xn, int C, int Ntok) {
    __shared__ ushort_t tile[32][33];
    int c0 = blockIdx.x * 32, n0 = blockIdx.y * 32, b = blockIdx.z;
    int tid = threadIdx.x;
    const float* xb = x + (long)b * C * Ntok;
    #pragma unroll
    for (int p = 0; p < 4; ++p) {
        int cl = p * 8 + (tid >> 5);
        int nl = tid & 31;
        float v  = xb[(long)(c0 + cl) * Ntok + n0 + nl];
        long  tk = (long)b * Ntok + n0 + nl;
        float mu = stats[2 * tk], rs = stats[2 * tk + 1];
        float y  = (v - mu) * rs * gamma[c0 + cl] + beta[c0 + cl];
        tile[nl][cl] = f2bf(y);
    }
    __syncthreads();
    ushort_t* Xb = Xn + (long)b * Ntok * C;
    #pragma unroll
    for (int p = 0; p < 4; ++p) {
        int nl = p * 8 + (tid >> 5);
        int cl = tid & 31;
        Xb[(long)(n0 + nl) * C + c0 + cl] = tile[nl][cl];
    }
}

// ------------- V transpose: qkv[b][m][2C+d] -> Vt[b][d][m] (bf16) -------------
__global__ void transpose_v_kernel(const ushort_t* __restrict__ qkv,
                                   ushort_t* __restrict__ Vt, int C, int Ntok) {
    __shared__ ushort_t t[32][33];
    int d0 = blockIdx.x * 32, m0 = blockIdx.y * 32, b = blockIdx.z;
    int tid = threadIdx.x;
    const ushort_t* src = qkv + (long)b * Ntok * 3 * C + 2 * C;
    #pragma unroll
    for (int p = 0; p < 4; ++p) {
        int ml = p * 8 + (tid >> 5);
        int dl = tid & 31;
        t[dl][ml] = src[(long)(m0 + ml) * 3 * C + d0 + dl];
    }
    __syncthreads();
    ushort_t* dst = Vt + (long)b * C * Ntok;
    #pragma unroll
    for (int p = 0; p < 4; ++p) {
        int dl = p * 8 + (tid >> 5);
        int ml = tid & 31;
        dst[(long)(d0 + dl) * Ntok + m0 + ml] = t[dl][ml];
    }
}

// ---------------- row softmax: S(f32, 4096 wide) -> P(bf16) ----------------
__global__ void softmax_kernel(const float* __restrict__ S,
                               ushort_t* __restrict__ P, int Ntok) {
    int n = blockIdx.x, b = blockIdx.z;
    const float* row  = S + ((long)b * Ntok + n) * Ntok;
    ushort_t*    prow = P + ((long)b * Ntok + n) * Ntok;
    int tid = threadIdx.x;                   // 256 threads, 16 elems each
    float vals[16];
    float m = -INFINITY;
    #pragma unroll
    for (int i = 0; i < 16; ++i) { vals[i] = row[i * 256 + tid]; m = fmaxf(m, vals[i]); }
    __shared__ float red[256];
    red[tid] = m; __syncthreads();
    for (int s = 128; s > 0; s >>= 1) { if (tid < s) red[tid] = fmaxf(red[tid], red[tid + s]); __syncthreads(); }
    m = red[0]; __syncthreads();
    float sum = 0.f;
    #pragma unroll
    for (int i = 0; i < 16; ++i) { vals[i] = __expf(vals[i] - m); sum += vals[i]; }
    red[tid] = sum; __syncthreads();
    for (int s = 128; s > 0; s >>= 1) { if (tid < s) red[tid] += red[tid + s]; __syncthreads(); }
    float inv = 1.0f / red[0];
    #pragma unroll
    for (int i = 0; i < 16; ++i) prow[i * 256 + tid] = f2bf(vals[i] * inv);
}

// ---------------- bf16 WMMA GEMM:  C[M][N] = A[M][K] * B[N][K]^T ----------------
// 256 threads = 8 waves arranged 4(M) x 2(N); block tile 128x128; wave tile 32x64
// (2x4 grid of 16x16 wmma frags). Double-buffered LDS staged with async-to-LDS.
// EP=0: store bf16 C[row*ldc+col]
// EP=1: store f32  C[row*ldc+col] * scale
// EP=2: store f32  C[col*ldc+row] + bias[col]   (transposed epilogue)
template<int EP>
__global__ void gemm_kernel(const ushort_t* __restrict__ A, int lda, long sAb,
                            const ushort_t* __restrict__ Bmat, int ldb, long sBb,
                            void* __restrict__ Cv, int ldc, long sCb,
                            int K, float scale, const float* __restrict__ bias) {
    __shared__ __align__(16) ushort_t As[2][BM][LDP];   // 2 x 128 x 80B = 20 KB
    __shared__ __align__(16) ushort_t Bs[2][BN][LDP];   // 2 x 128 x 80B = 20 KB

    const int tid  = threadIdx.x;
    const int lane = tid & 31;
    const int wave = tid >> 5;
    const int sub  = lane & 15;
    const int half = lane >> 4;
    const int waveM = (wave & 3) * 32;
    const int waveN = (wave >> 2) * 64;
    const int mBase = blockIdx.y * BM;
    const int nBase = blockIdx.x * BN;
    const int b = blockIdx.z;
    const ushort_t* Ab = A    + (long)b * sAb;
    const ushort_t* Bb = Bmat + (long)b * sBb;

    // per-thread copy slots: tile 128 rows x 32 bf16 (64B) = 512 x 16B chunks
    const int r0 = tid >> 2, c0 = (tid & 3) * 8;        // slot for i=0
    const int r1 = (tid + 256) >> 2, c1 = ((tid + 256) & 3) * 8;

    auto issue_tile = [&](int kt, int buf) {
        int k0 = kt * BK;
        copy16_g2l(Ab + (long)(mBase + r0) * lda + k0 + c0, &As[buf][r0][c0]);
        copy16_g2l(Ab + (long)(mBase + r1) * lda + k0 + c1, &As[buf][r1][c1]);
        copy16_g2l(Bb + (long)(nBase + r0) * ldb + k0 + c0, &Bs[buf][r0][c0]);
        copy16_g2l(Bb + (long)(nBase + r1) * ldb + k0 + c1, &Bs[buf][r1][c1]);
    };

    v8f acc[2][4] = {};
    union Frag { v16bf v; uint4 q[2]; };

    const int nK = K / BK;
    issue_tile(0, 0);
    for (int kt = 0; kt < nK; ++kt) {
        const int buf = kt & 1;
        async_wait_all();
        __syncthreads();
        if (kt + 1 < nK) issue_tile(kt + 1, buf ^ 1);

        // Fragments per ISA 7.12.2: A 16x32 (k = half*8 + pairs, +16 in upper quad),
        // B 32x16 (same k order as A). Both are two b128 LDS loads.
        Frag af[2], bfr[4];
        #pragma unroll
        for (int mi = 0; mi < 2; ++mi) {
            const uint4* ar = (const uint4*)(&As[buf][waveM + mi * 16 + sub][0]);
            af[mi].q[0] = ar[half];
            af[mi].q[1] = ar[half + 2];
        }
        #pragma unroll
        for (int ni = 0; ni < 4; ++ni) {
            const uint4* br = (const uint4*)(&Bs[buf][waveN + ni * 16 + sub][0]);
            bfr[ni].q[0] = br[half];
            bfr[ni].q[1] = br[half + 2];
        }
        #pragma unroll
        for (int mi = 0; mi < 2; ++mi)
            #pragma unroll
            for (int ni = 0; ni < 4; ++ni)
                acc[mi][ni] = __builtin_amdgcn_wmma_f32_16x16x32_bf16(
                    false, af[mi].v, false, bfr[ni].v,
                    (short)0, acc[mi][ni], false, false);
        __syncthreads();
    }

    // Epilogue: C/D layout -> element (m = r + 8*half, n = sub) for acc vgpr r.
    #pragma unroll
    for (int mi = 0; mi < 2; ++mi)
        #pragma unroll
        for (int ni = 0; ni < 4; ++ni)
            #pragma unroll
            for (int r = 0; r < 8; ++r) {
                int row = mBase + waveM + mi * 16 + half * 8 + r;
                int col = nBase + waveN + ni * 16 + sub;
                float v = acc[mi][ni][r];
                if (EP == 0) {
                    ((ushort_t*)Cv)[(long)b * sCb + (long)row * ldc + col] = f2bf(v);
                } else if (EP == 1) {
                    ((float*)Cv)[(long)b * sCb + (long)row * ldc + col] = v * scale;
                } else {
                    ((float*)Cv)[(long)b * sCb + (long)col * ldc + row] = v + bias[col];
                }
            }
}

// ---------------- host orchestration ----------------
extern "C" void kernel_launch(void* const* d_in, const int* in_sizes, int n_in,
                              void* d_out, int out_size, void* d_ws, size_t ws_size,
                              hipStream_t stream) {
    const int Bb = 2, C = 512, Ntok = 4096;

    const float* x      = (const float*)d_in[0];
    const float* gamma  = (const float*)d_in[1];
    const float* beta   = (const float*)d_in[2];
    const float* w_qkv  = (const float*)d_in[3];
    const float* w_proj = (const float*)d_in[4];
    const float* b_proj = (const float*)d_in[5];
    float* out = (float*)d_out;

    char* ws = (char*)d_ws;
    size_t off = 0;
    auto alloc = [&](size_t bytes) -> void* {
        off = (off + 255) & ~(size_t)255;
        void* p = ws + off;
        off += bytes;
        return p;
    };

    float*    stats = (float*)   alloc((size_t)Bb * Ntok * 2 * sizeof(float));
    ushort_t* Xn    = (ushort_t*)alloc((size_t)Bb * Ntok * C * 2);
    ushort_t* Wqkv  = (ushort_t*)alloc((size_t)3 * C * C * 2);
    ushort_t* Wproj = (ushort_t*)alloc((size_t)C * C * 2);
    ushort_t* qkv   = (ushort_t*)alloc((size_t)Bb * Ntok * 3 * C * 2);
    ushort_t* Vt    = (ushort_t*)alloc((size_t)Bb * C * Ntok * 2);
    float*    S     = (float*)   alloc((size_t)Bb * Ntok * Ntok * 4);
    ushort_t* P     = (ushort_t*)alloc((size_t)Bb * Ntok * Ntok * 2);
    ushort_t* O     = (ushort_t*)alloc((size_t)Bb * Ntok * C * 2);

    // 1. weights -> bf16
    cvt_f32_bf16_kernel<<<(3 * C * C + 255) / 256, 256, 0, stream>>>(w_qkv, Wqkv, 3 * C * C);
    cvt_f32_bf16_kernel<<<(C * C + 255) / 256, 256, 0, stream>>>(w_proj, Wproj, C * C);

    // 2. LayerNorm
    ln_stats_kernel<<<(Bb * Ntok) / 256, 256, 0, stream>>>(x, stats, C, Ntok);
    ln_apply_kernel<<<dim3(C / 32, Ntok / 32, Bb), 256, 0, stream>>>(
        x, stats, gamma, beta, Xn, C, Ntok);

    // 3. QKV = Xn * Wqkv^T  -> bf16 [B][N][3C]
    gemm_kernel<0><<<dim3((3 * C) / BN, Ntok / BM, Bb), 256, 0, stream>>>(
        Xn, C, (long)Ntok * C, Wqkv, C, 0,
        qkv, 3 * C, (long)Ntok * 3 * C, C, 1.f, nullptr);

    // 4. Vt[b][d][m] = V^T (so the P*V GEMM can use the fast K-major path)
    transpose_v_kernel<<<dim3(C / 32, Ntok / 32, Bb), 256, 0, stream>>>(qkv, Vt, C, Ntok);

    // 5. S = (Q * K^T) * 1/sqrt(C)  -> f32 [B][N][N]
    gemm_kernel<1><<<dim3(Ntok / BN, Ntok / BM, Bb), 256, 0, stream>>>(
        qkv + 0, 3 * C, (long)Ntok * 3 * C,
        qkv + C, 3 * C, (long)Ntok * 3 * C,
        S, Ntok, (long)Ntok * Ntok, C, 1.0f / sqrtf((float)C), nullptr);

    // 6. P = softmax(S) -> bf16
    softmax_kernel<<<dim3(Ntok, 1, Bb), 256, 0, stream>>>(S, P, Ntok);

    // 7. O = P * Vt^T -> bf16 [B][N][C]
    gemm_kernel<0><<<dim3(C / BN, Ntok / BM, Bb), 256, 0, stream>>>(
        P, Ntok, (long)Ntok * Ntok,
        Vt, Ntok, (long)C * Ntok,
        O, C, (long)Ntok * C, Ntok, 1.f, nullptr);

    // 8. out = O * Wproj^T + b, stored transposed into [B][C][N] (= BCTHW)
    gemm_kernel<2><<<dim3(C / BN, Ntok / BM, Bb), 256, 0, stream>>>(
        O, C, (long)Ntok * C, Wproj, C, 0,
        out, Ntok, (long)C * Ntok, C, 1.f, b_proj);

    (void)in_sizes; (void)n_in; (void)out_size; (void)ws_size;
}
